// CausalSelfAttention_687194767434
// MI455X (gfx1250) — matrix operand order repaired and checked by reading, not compiled, not run on hardware
//
#include <hip/hip_runtime.h>

// ---------- types ----------
typedef __attribute__((ext_vector_type(16))) _Float16 v16h;
typedef __attribute__((ext_vector_type(8)))  _Float16 v8h;
typedef __attribute__((ext_vector_type(8)))  float    v8f;

union V16 { v16h v; v8h h[2]; };

__device__ __forceinline__ v8f wmma_f16(v16h a, v16h b, v8f c) {
  // D = A(16x32 f16) * B(32x16 f16) + C(16x16 f32)
  return __builtin_amdgcn_wmma_f32_16x16x32_f16(false, a, false, b, (short)0, c,
                                                false, false);
}

// ---- CDNA5 async global->LDS copy (ASYNCcnt-tracked, bypasses VGPRs) ----
// GV mode: vdst = LDS byte offset VGPR, vaddr = 64-bit global address VGPRs.
// LDS byte offset = low 32 bits of the flat shared-aperture pointer.
__device__ __forceinline__ void async_copy16(void* lds, const void* g) {
  asm volatile("global_load_async_to_lds_b128 %0, %1, off"
               :: "v"((unsigned)(unsigned long long)lds),
                  "v"((unsigned long long)g)
               : "memory");
}
__device__ __forceinline__ void wait_async0() {
  asm volatile("s_wait_asynccnt 0" ::: "memory");
}

// ---------- problem constants ----------
constexpr int Bn = 8, Tn = 1024, Cn = 768, Hn = 12, Dn = 64;
constexpr int M1 = Bn * Tn;   // 8192 rows (B*T)
constexpr int N1 = 3 * Cn;    // 2304 qkv cols
constexpr int KK = Cn;        // 768 reduction dim (both GEMMs)

// =====================================================================
// prep: f32 -> f16 convert
// =====================================================================
__global__ void k_f32_to_f16(const float* __restrict__ src,
                             _Float16* __restrict__ dst, int n) {
  int i = blockIdx.x * blockDim.x + threadIdx.x;
  if (i < n) dst[i] = (_Float16)src[i];
}

// prep: src[K][N] f32 -> dst[N][K] f16   (weights to N-major for B frags)
__global__ void k_transpose_to_f16(const float* __restrict__ src,
                                   _Float16* __restrict__ dst, int K, int N) {
  int i = blockIdx.x * blockDim.x + threadIdx.x;
  if (i < K * N) {
    int n = i / K;
    int k = i - n * K;
    dst[i] = (_Float16)src[k * N + n];
  }
}

// =====================================================================
// register-blocked 64x64-per-wave GEMM core: 16 wmma per 32-K step
//   A row-major [M][K] f16, B stored N-major [N][K] f16
// =====================================================================
__device__ __forceinline__ void gemm64x64(const _Float16* __restrict__ A,
                                          const _Float16* __restrict__ Bm,
                                          int lr, int half, v8f acc[4][4]) {
  const _Float16* ar[4];
  const _Float16* br[4];
#pragma unroll
  for (int i = 0; i < 4; ++i) {
    ar[i] = A  + (size_t)(i * 16 + lr) * KK;
    br[i] = Bm + (size_t)(i * 16 + lr) * KK;
  }
#pragma unroll 2
  for (int k = 0; k < KK; k += 32) {
    V16 af[4], bf[4];
#pragma unroll
    for (int i = 0; i < 4; ++i) {
      // A 16x32 frag: lane M = lane&15; K 0..15 (lanes 0-15) / 16..31 (16-31)
      af[i].h[0] = *(const v8h*)(ar[i] + k + half * 16);
      af[i].h[1] = *(const v8h*)(ar[i] + k + half * 16 + 8);
      // B 32x16 frag: lane N = lane&15; K 0..15 (lanes 0-15) / 16..31 (16-31)
      bf[i].h[0] = *(const v8h*)(br[i] + k + half * 16);
      bf[i].h[1] = *(const v8h*)(br[i] + k + half * 16 + 8);
    }
#pragma unroll
    for (int i = 0; i < 4; ++i)
#pragma unroll
      for (int j = 0; j < 4; ++j)
        acc[i][j] = wmma_f16(af[i].v, bf[j].v, acc[i][j]);
  }
}

// =====================================================================
// GEMM 1: qkv = x @ Wqkv + b ; scatter into attention layouts
//   qh,kh : [B*H][T][64]   vt : [B*H][64][T]   (q pre-scaled by 1/sqrt(D))
//   block = 8 waves (2x4) -> 128 x 256 tile
// =====================================================================
__global__ __launch_bounds__(256)
void k_gemm_qkv(const _Float16* __restrict__ xh,
                const _Float16* __restrict__ wt,   // [2304][768]
                const float* __restrict__ bias,    // [2304]
                _Float16* __restrict__ qh,
                _Float16* __restrict__ kh,
                _Float16* __restrict__ vt) {
  const int lane = threadIdx.x & 31, wave = threadIdx.x >> 5;
  const int lr = lane & 15, half = lane >> 4;
  const int wm = wave >> 2, wn = wave & 3;
  const int m0 = blockIdx.y * 128 + wm * 64;
  const int n0 = blockIdx.x * 256 + wn * 64;

  v8f acc[4][4];
  v8f zero = {};
#pragma unroll
  for (int i = 0; i < 4; ++i)
#pragma unroll
    for (int j = 0; j < 4; ++j) acc[i][j] = zero;

  gemm64x64(xh + (size_t)m0 * KK, wt + (size_t)n0 * KK, lr, half, acc);

#pragma unroll
  for (int j = 0; j < 4; ++j) {
    int n = n0 + j * 16 + lr;            // output column (fixed per lane)
    float bv = bias[n];
    int which = n / Cn;                  // 0=q 1=k 2=v (uniform per wave)
    int c = n - which * Cn;
    int h = c >> 6, d = c & 63;
#pragma unroll
    for (int i = 0; i < 4; ++i) {
#pragma unroll
      for (int r = 0; r < 8; ++r) {
        int m = m0 + i * 16 + r + half * 8;     // row = b*T + t
        int b_ = m >> 10, t = m & (Tn - 1);
        int bh = b_ * Hn + h;
        float val = acc[i][j][r] + bv;
        if (which == 0) {
          qh[((size_t)bh * Tn + t) * Dn + d] = (_Float16)(val * 0.125f);
        } else if (which == 1) {
          kh[((size_t)bh * Tn + t) * Dn + d] = (_Float16)val;
        } else {
          vt[((size_t)bh * Dn + d) * Tn + t] = (_Float16)val;
        }
      }
    }
  }
}

// =====================================================================
// Flash attention: block = 128 query rows (8 waves x 16).  K/V tiles are
// double-buffered in LDS via async global->LDS copies (ASYNCcnt), so the
// next tile's DMA overlaps the current tile's 8 WMMAs + softmax; one
// s_wait_asynccnt + one barrier per 32-key step.
// =====================================================================
__global__ __launch_bounds__(256)
void k_attention(const _Float16* __restrict__ qh,
                 const _Float16* __restrict__ kh,
                 const _Float16* __restrict__ vt,
                 _Float16* __restrict__ yh) {   // [B*T][C]
  __shared__ _Float16 Ksh[2][32][80];  // [buf][key][d] row=160B (16B aligned)
  __shared__ _Float16 Vsh[2][64][40];  // [buf][d][key] row= 80B (16B aligned)
  __shared__ _Float16 pt[8][16][32];   // per-wave P transpose tile

  const int tid = threadIdx.x;
  const int lane = tid & 31, wave = tid >> 5;
  const int lr = lane & 15, half = lane >> 4;
  const int bh = blockIdx.y;
  const int b = bh / Hn, h = bh - b * Hn;
  const int qB = blockIdx.x * 128;          // first query row of this block
  const int qb = qB + wave * 16;            // first query row of this wave

  const _Float16* Q = qh + (size_t)bh * Tn * Dn;
  const _Float16* K = kh + (size_t)bh * Tn * Dn;
  const _Float16* V = vt + (size_t)bh * Dn * Tn;

  // cooperative async staging of one 32-key K/V tile pair into buffer `buf`
  const int skey = tid >> 3, sch = (tid & 7) * 8;   // K: 32 rows x 128B
  const int sd   = tid >> 2, svch = (tid & 3) * 8;  // V: 64 rows x  64B
  auto stage = [&](int s, int buf) {
    const int k0 = s * 32;
    async_copy16(&Ksh[buf][skey][sch], K + (size_t)(k0 + skey) * Dn + sch);
    async_copy16(&Vsh[buf][sd][svch], V + (size_t)sd * Tn + k0 + svch);
  };

  // Q A-fragments (d 0..31 and 32..63), resident for the whole loop
  V16 qa0, qa1;
  {
    const _Float16* qrow = Q + (size_t)(qb + lr) * Dn;
    qa0.h[0] = *(const v8h*)(qrow + half * 16);
    qa0.h[1] = *(const v8h*)(qrow + half * 16 + 8);
    qa1.h[0] = *(const v8h*)(qrow + 32 + half * 16);
    qa1.h[1] = *(const v8h*)(qrow + 32 + half * 16 + 8);
  }

  v8f acc0 = {}, acc1 = {}, acc2 = {}, acc3 = {};
  float mrow[8], lrow[8];
#pragma unroll
  for (int r = 0; r < 8; ++r) { mrow[r] = -3.0e38f; lrow[r] = 0.f; }

  const int nstepsBlk = (qB + 128) >> 5;      // block-uniform causal bound
  const int myend     = (qb + 16 + 31) >> 5;  // this wave's causal bound

  stage(0, 0);
  wait_async0();
  __syncthreads();

  for (int s = 0; s < nstepsBlk; ++s) {
    const int k0 = s * 32;
    const int buf = s & 1;

    // prefetch next K/V tile into the other buffer (overlaps compute)
    if (s + 1 < nstepsBlk) stage(s + 1, buf ^ 1);

    if (s < myend) {
      // ---- scores: S[16 x 32] = Q @ K^T (scale folded into Q) ----
      v8f s0 = {}, s1 = {};
      {
        V16 f;
        f.h[0] = *(const v8h*)&Ksh[buf][lr][half * 16];
        f.h[1] = *(const v8h*)&Ksh[buf][lr][half * 16 + 8];
        s0 = wmma_f16(qa0.v, f.v, s0);
        f.h[0] = *(const v8h*)&Ksh[buf][lr][32 + half * 16];
        f.h[1] = *(const v8h*)&Ksh[buf][lr][32 + half * 16 + 8];
        s0 = wmma_f16(qa1.v, f.v, s0);
        f.h[0] = *(const v8h*)&Ksh[buf][16 + lr][half * 16];
        f.h[1] = *(const v8h*)&Ksh[buf][16 + lr][half * 16 + 8];
        s1 = wmma_f16(qa0.v, f.v, s1);
        f.h[0] = *(const v8h*)&Ksh[buf][16 + lr][32 + half * 16];
        f.h[1] = *(const v8h*)&Ksh[buf][16 + lr][32 + half * 16 + 8];
        s1 = wmma_f16(qa1.v, f.v, s1);
      }

      // ---- causal mask + online softmax (16-lane half reductions) ----
      float p0[8], p1[8];
#pragma unroll
      for (int r = 0; r < 8; ++r) {
        int m = qb + r + half * 8;
        float x0 = ((k0 + lr) <= m) ? s0[r] : -3.0e38f;
        float x1 = ((k0 + 16 + lr) <= m) ? s1[r] : -3.0e38f;
        float mx = fmaxf(x0, x1);
        mx = fmaxf(mx, __shfl_xor(mx, 1, 32));
        mx = fmaxf(mx, __shfl_xor(mx, 2, 32));
        mx = fmaxf(mx, __shfl_xor(mx, 4, 32));
        mx = fmaxf(mx, __shfl_xor(mx, 8, 32));
        float mnew = fmaxf(mrow[r], mx);
        float alpha = __expf(mrow[r] - mnew);
        mrow[r] = mnew;
        float e0 = __expf(x0 - mnew);
        float e1 = __expf(x1 - mnew);
        p0[r] = e0; p1[r] = e1;
        float rs = e0 + e1;
        rs += __shfl_xor(rs, 1, 32);
        rs += __shfl_xor(rs, 2, 32);
        rs += __shfl_xor(rs, 4, 32);
        rs += __shfl_xor(rs, 8, 32);
        lrow[r] = lrow[r] * alpha + rs;
        acc0[r] *= alpha; acc1[r] *= alpha; acc2[r] *= alpha; acc3[r] *= alpha;
      }

      // ---- P: C-layout -> A-layout via per-wave LDS tile ----
#pragma unroll
      for (int r = 0; r < 8; ++r) {
        pt[wave][r + half * 8][lr]      = (_Float16)p0[r];
        pt[wave][r + half * 8][lr + 16] = (_Float16)p1[r];
      }
      V16 pa;
      pa.h[0] = *(const v8h*)(&pt[wave][lr][half * 16]);
      pa.h[1] = *(const v8h*)(&pt[wave][lr][half * 16 + 8]);

      // ---- acc += P(16x32) @ V(32x64) : 4 wmma over d-tiles ----
      {
        V16 vb;
        vb.h[0] = *(const v8h*)&Vsh[buf][lr][half * 16];
        vb.h[1] = *(const v8h*)&Vsh[buf][lr][half * 16 + 8];
        acc0 = wmma_f16(pa.v, vb.v, acc0);
        vb.h[0] = *(const v8h*)&Vsh[buf][16 + lr][half * 16];
        vb.h[1] = *(const v8h*)&Vsh[buf][16 + lr][half * 16 + 8];
        acc1 = wmma_f16(pa.v, vb.v, acc1);
        vb.h[0] = *(const v8h*)&Vsh[buf][32 + lr][half * 16];
        vb.h[1] = *(const v8h*)&Vsh[buf][32 + lr][half * 16 + 8];
        acc2 = wmma_f16(pa.v, vb.v, acc2);
        vb.h[0] = *(const v8h*)&Vsh[buf][48 + lr][half * 16];
        vb.h[1] = *(const v8h*)&Vsh[buf][48 + lr][half * 16 + 8];
        acc3 = wmma_f16(pa.v, vb.v, acc3);
      }
    }

    wait_async0();     // next tile's DMA complete (wave-local)
    __syncthreads();   // all waves done reading buf / writing buf^1
  }

  // ---- finalize: divide by row sums, store y [B*T][C] as f16 ----
#pragma unroll
  for (int r = 0; r < 8; ++r) {
    float inv = 1.0f / lrow[r];
    int t = qb + r + half * 8;
    _Float16* yr = yh + (size_t)(b * Tn + t) * Cn + h * Dn;
    yr[lr]      = (_Float16)(acc0[r] * inv);
    yr[16 + lr] = (_Float16)(acc1[r] * inv);
    yr[32 + lr] = (_Float16)(acc2[r] * inv);
    yr[48 + lr] = (_Float16)(acc3[r] * inv);
  }
}

// =====================================================================
// GEMM 2: out = y @ Wproj + b  (f32 output), 64x64 per wave
// =====================================================================
__global__ __launch_bounds__(256)
void k_gemm_proj(const _Float16* __restrict__ yh,
                 const _Float16* __restrict__ wt,   // [768][768]
                 const float* __restrict__ bias,
                 float* __restrict__ out) {
  const int lane = threadIdx.x & 31, wave = threadIdx.x >> 5;
  const int lr = lane & 15, half = lane >> 4;
  const int wm = wave >> 2, wn = wave & 3;
  const int m0 = blockIdx.y * 128 + wm * 64;
  const int n0 = blockIdx.x * 256 + wn * 64;

  v8f acc[4][4];
  v8f zero = {};
#pragma unroll
  for (int i = 0; i < 4; ++i)
#pragma unroll
    for (int j = 0; j < 4; ++j) acc[i][j] = zero;

  gemm64x64(yh + (size_t)m0 * KK, wt + (size_t)n0 * KK, lr, half, acc);

#pragma unroll
  for (int j = 0; j < 4; ++j) {
    int n = n0 + j * 16 + lr;
    float bv = bias[n];
#pragma unroll
    for (int i = 0; i < 4; ++i) {
#pragma unroll
      for (int r = 0; r < 8; ++r) {
        int m = m0 + i * 16 + r + half * 8;
        out[(size_t)m * Cn + n] = acc[i][j][r] + bv;
      }
    }
  }
}

// =====================================================================
// host launcher
// =====================================================================
extern "C" void kernel_launch(void* const* d_in, const int* in_sizes, int n_in,
                              void* d_out, int out_size, void* d_ws,
                              size_t ws_size, hipStream_t stream) {
  const float* x     = (const float*)d_in[0];
  const float* wattn = (const float*)d_in[1];
  const float* battn = (const float*)d_in[2];
  const float* wproj = (const float*)d_in[3];
  const float* bproj = (const float*)d_in[4];
  float* out = (float*)d_out;

  // workspace carve-up (all f16), total ~68 MB
  char* ws = (char*)d_ws;
  size_t off = 0;
  auto carve = [&](size_t elems) {
    void* p = ws + off;
    off = (off + elems * sizeof(_Float16) + 255) & ~(size_t)255;
    return (_Float16*)p;
  };
  _Float16* xh    = carve((size_t)M1 * KK);       // x in f16
  _Float16* wqkvt = carve((size_t)N1 * KK);       // Wqkv^T  [2304][768]
  _Float16* wpt   = carve((size_t)Cn * KK);       // Wproj^T [768][768]
  _Float16* qh    = carve((size_t)Bn * Hn * Tn * Dn);
  _Float16* kh    = carve((size_t)Bn * Hn * Tn * Dn);
  _Float16* vt    = carve((size_t)Bn * Hn * Dn * Tn);
  _Float16* yh    = carve((size_t)M1 * Cn);

  // prep
  k_f32_to_f16<<<(M1 * KK + 255) / 256, 256, 0, stream>>>(x, xh, M1 * KK);
  k_transpose_to_f16<<<(N1 * KK + 255) / 256, 256, 0, stream>>>(wattn, wqkvt,
                                                                KK, N1);
  k_transpose_to_f16<<<(Cn * KK + 255) / 256, 256, 0, stream>>>(wproj, wpt,
                                                                KK, Cn);
  // qkv projection: block tile 128x256
  dim3 g1(N1 / 256, M1 / 128);
  k_gemm_qkv<<<g1, 256, 0, stream>>>(xh, wqkvt, battn, qh, kh, vt);
  // attention: grid (T/128 query blocks, B*H heads)
  dim3 ga(Tn / 128, Bn * Hn);
  k_attention<<<ga, 256, 0, stream>>>(qh, kh, vt, yh);
  // output projection: block tile 128x256
  dim3 g2(Cn / 256, M1 / 128);
  k_gemm_proj<<<g2, 256, 0, stream>>>(yh, wpt, bproj, out);
}
